// InfiniteContextAttention_8761733284163
// MI455X (gfx1250) — matrix operand order repaired and checked by reading, not compiled, not run on hardware
//
#include <hip/hip_runtime.h>
#include <hip/hip_bf16.h>

// ---------------------------------------------------------------------------
// CDNA5 (gfx1250, wave32) WMMA types
// ---------------------------------------------------------------------------
typedef __attribute__((ext_vector_type(16))) __bf16 v16bf;
typedef __attribute__((ext_vector_type(8)))  float  v8f;

// A-fragment (16x32 bf16, M x K), ISA 7.12.2 layout:
//   lane l (m = l&15, lg = l>>4): elems 0..7  = K (8*lg + 0..7)
//                                 elems 8..15 = K (16 + 8*lg + 0..7)
// `base` must already point at row m, offset kb + 8*lg.
static __device__ inline v16bf load_a_frag_f32(const float* __restrict__ base) {
  v16bf a;
#pragma unroll
  for (int e = 0; e < 8; ++e) a[e] = (__bf16)base[e];
#pragma unroll
  for (int e = 0; e < 8; ++e) a[8 + e] = (__bf16)base[16 + e];
  return a;
}

// B-fragment (32x16 bf16, K x N): lane l (n = l&15, lg = l>>4) holds
// elems 0..7 = K (8*lg + 0..7), elems 8..15 = K (16 + 8*lg + 0..7). For NT GEMM, column n of B == row n of W.
static __device__ inline v16bf load_b_frag_f32(const float* __restrict__ base) {
  v16bf b;
#pragma unroll
  for (int e = 0; e < 16; ++e) b[e] = (__bf16)base[(e & 7) + ((e >> 3) << 4)];
  return b;
}

// ---------------------------------------------------------------------------
// Split-K NT GEMM: OUT_part(ks) = X(32xK) * W(NxK)^T over K-chunk ks.
// Weight streaming is the roofline (~67MB/GEMM -> ~3us at 23.3TB/s); split-K
// x4 gives 2048 wave-tiles so enough loads are in flight to saturate HBM.
// K-loop unrolled x2: 2 wmmas + 2x(8 b128 loads) per backedge.
// grid.x = 64 * nsplit blocks, 8 waves/block.
// ---------------------------------------------------------------------------
__global__ __launch_bounds__(256) void gemm_nt_32_sk(const float* __restrict__ X,
                                                     const float* __restrict__ W,
                                                     float* __restrict__ OUT,
                                                     int nsplit) {
  constexpr int K = 4096, NCOL = 4096;
  const int lane = threadIdx.x & 31;
  const int wave = threadIdx.x >> 5;
  int tile = blockIdx.x * 8 + wave;
  const int mt = tile & 1;  tile >>= 1;   // 2 M-tiles (32 rows)
  const int nt = tile & 255; tile >>= 8;  // 256 N-tiles
  const int ks = tile;                    // split-K slice
  const int lg = lane >> 4;               // lane group
  const int li = lane & 15;

  const int kchunk = K / nsplit;
  const int kb0 = ks * kchunk;

  const float* __restrict__ xk = X + (size_t)(mt * 16 + li) * K + kb0;
  const float* __restrict__ wk = W + (size_t)(nt * 16 + li) * K + kb0;

  v8f acc = {};
  for (int kb = 0; kb < kchunk; kb += 64) {
    __builtin_prefetch(wk + kb + 1024, 0, 0);   // global_prefetch on streamed W
    v16bf a0 = load_a_frag_f32(xk + kb + 8 * lg);
    v16bf b0 = load_b_frag_f32(wk + kb + 8 * lg);
    v16bf a1 = load_a_frag_f32(xk + kb + 32 + 8 * lg);
    v16bf b1 = load_b_frag_f32(wk + kb + 32 + 8 * lg);
    acc = __builtin_amdgcn_wmma_f32_16x16x32_bf16(false, a0, false, b0,
                                                  (short)0, acc, false, false);
    acc = __builtin_amdgcn_wmma_f32_16x16x32_bf16(false, a1, false, b1,
                                                  (short)0, acc, false, false);
  }

  // C/D layout: lanes 0-15 hold M = r, lanes 16-31 hold M = 8 + r; N = li.
  float* __restrict__ o = OUT + (size_t)ks * 32 * NCOL;
#pragma unroll
  for (int r = 0; r < 8; ++r) {
    int row = mt * 16 + r + 8 * lg;
    o[(size_t)row * NCOL + nt * 16 + li] = acc[r];
  }
}

// Sum nsplit partials (32x4096 each) -> out. float4 per thread, grid 128x256.
__global__ __launch_bounds__(256) void reduce_sk(const float* __restrict__ part,
                                                 float* __restrict__ out,
                                                 int nsplit) {
  const int i = (blockIdx.x * 256 + threadIdx.x) * 4;
  float4 s = *(const float4*)(part + i);
  for (int ks = 1; ks < nsplit; ++ks) {
    float4 p = *(const float4*)(part + (size_t)ks * 131072 + i);
    s.x += p.x; s.y += p.y; s.z += p.z; s.w += p.w;
  }
  *(float4*)(out + i) = s;
}

// ---------------------------------------------------------------------------
// Attention: under the reference's causal mask (key index s <= query row i,
// i < 16) only positions 0..15 of full_k survive == compressed_k_in[:,:,:16].
// One wave per (b,h):  S = softmax(mask(q.ck^T/sqrt(128))),  O = S.cv
// ---------------------------------------------------------------------------
__global__ __launch_bounds__(32) void attn16(const float* __restrict__ Q,
                                             const float* __restrict__ CK,
                                             const float* __restrict__ CV,
                                             float* __restrict__ OUTWS) {
  constexpr int HID = 4096, D = 128, CL = 2048, H = 32;
  __shared__ float sattn[16 * 16];

  const int lane = threadIdx.x & 31;
  const int lg = lane >> 4;
  const int li = lane & 15;
  const int b = blockIdx.x >> 5;  // H = 32
  const int h = blockIdx.x & 31;

  const float* __restrict__ qbase  = Q  + (size_t)(b * 16) * HID + h * D;
  const float* __restrict__ ckbase = CK + (size_t)(b * H + h) * CL * D;
  const float* __restrict__ cvbase = CV + (size_t)(b * H + h) * CL * D;

  // scores = q (16xD) . ck^T : A rows = q rows, B cols = keys j (K-contig)
  v8f acc = {};
#pragma unroll
  for (int kb = 0; kb < D; kb += 32) {
    v16bf a = load_a_frag_f32(qbase + (size_t)li * HID + kb + 8 * lg);
    v16bf k = load_b_frag_f32(ckbase + (size_t)li * D + kb + 8 * lg);
    acc = __builtin_amdgcn_wmma_f32_16x16x32_bf16(false, a, false, k,
                                                  (short)0, acc, false, false);
  }

  // Causal mask + softmax. Row m = r + 8*lg spans the 16-lane group (N = li):
  // width-16 shuffles are the wave32-correct reduction.
  const float scale = 0.08838834764831845f;  // 1/sqrt(128)
  float p[8];
#pragma unroll
  for (int r = 0; r < 8; ++r) {
    const int mrow = r + 8 * lg;
    float s = acc[r] * scale;
    if (li > mrow) s = -3.0e38f;
    float mx = s;
#pragma unroll
    for (int off = 1; off < 16; off <<= 1) mx = fmaxf(mx, __shfl_xor(mx, off, 16));
    float e = __expf(s - mx);
    float sum = e;
#pragma unroll
    for (int off = 1; off < 16; off <<= 1) sum += __shfl_xor(sum, off, 16);
    p[r] = e / sum;
  }

  // Re-swizzle attn (C/D layout) -> A-fragment layout via LDS.
#pragma unroll
  for (int r = 0; r < 8; ++r) sattn[(r + 8 * lg) * 16 + li] = p[r];
  __syncthreads();

  // A-frag: row m = li, valid K = 0..15 (elems 0..7 = K 8*lg+e), K 16..31 = 0.
  v16bf pa;
#pragma unroll
  for (int e = 0; e < 8; ++e) pa[e] = (__bf16)sattn[li * 16 + 8 * lg + e];
#pragma unroll
  for (int e = 0; e < 8; ++e) pa[8 + e] = (__bf16)0.0f;

  // O = S . cv, 8 N-tiles of 16 over D=128.
#pragma unroll
  for (int t = 0; t < 8; ++t) {
    const int d0 = t * 16;
    v16bf pb;
#pragma unroll
    for (int e = 0; e < 16; ++e) {
      // B[k][n]: k = 8*lg + e for e < 8 (elems 8..15 are the zero K-padding 16..31)
      float v = (e < 8) ? cvbase[(size_t)(8 * lg + e) * D + d0 + li] : 0.0f;
      pb[e] = (__bf16)v;
    }
    v8f o = {};
    o = __builtin_amdgcn_wmma_f32_16x16x32_bf16(false, pa, false, pb,
                                                (short)0, o, false, false);
#pragma unroll
    for (int r = 0; r < 8; ++r) {
      const int mrow = r + 8 * lg;
      OUTWS[(size_t)(b * 16 + mrow) * HID + h * D + d0 + li] = o[r];
    }
  }
}

// ---------------------------------------------------------------------------
// Launch: q = hidden @ Wq^T ; attn ; out = attn_out @ Wo^T
// Wk/Wv/k_cache/v_cache are dead code under the reference's causal mask.
// Split-K x4 path when workspace permits (5MB), direct path otherwise.
// ---------------------------------------------------------------------------
extern "C" void kernel_launch(void* const* d_in, const int* in_sizes, int n_in,
                              void* d_out, int out_size, void* d_ws, size_t ws_size,
                              hipStream_t stream) {
  (void)in_sizes; (void)n_in; (void)out_size;
  const float* hidden = (const float*)d_in[0];
  const float* ck     = (const float*)d_in[3];
  const float* cv     = (const float*)d_in[4];
  const float* Wq     = (const float*)d_in[5];
  const float* Wo     = (const float*)d_in[8];
  float* out = (float*)d_out;

  constexpr int NSPLIT = 4;
  constexpr size_t ELTS = 32 * 4096;                         // 131072
  const size_t need_sk = (2 * NSPLIT + 2) * ELTS * sizeof(float);  // 5 MB

  if (ws_size >= need_sk) {
    float* qpart  = (float*)d_ws;                 // NSPLIT * ELTS
    float* q_ws   = qpart + (size_t)NSPLIT * ELTS;
    float* act_ws = q_ws + ELTS;
    float* apart  = act_ws + ELTS;                // NSPLIT * ELTS

    gemm_nt_32_sk<<<64 * NSPLIT, 256, 0, stream>>>(hidden, Wq, qpart, NSPLIT);
    reduce_sk<<<128, 256, 0, stream>>>(qpart, q_ws, NSPLIT);
    attn16<<<64, 32, 0, stream>>>(q_ws, ck, cv, act_ws);
    gemm_nt_32_sk<<<64 * NSPLIT, 256, 0, stream>>>(act_ws, Wo, apart, NSPLIT);
    reduce_sk<<<128, 256, 0, stream>>>(apart, out, NSPLIT);
  } else {
    float* q_ws   = (float*)d_ws;  // 512 KB
    float* act_ws = q_ws + ELTS;   // 512 KB
    gemm_nt_32_sk<<<64, 256, 0, stream>>>(hidden, Wq, q_ws, 1);
    attn16<<<64, 32, 0, stream>>>(q_ws, ck, cv, act_ws);
    gemm_nt_32_sk<<<64, 256, 0, stream>>>(act_ws, Wo, out, 1);
  }
}
